// SSMLayer_47579647705478
// MI455X (gfx1250) — matrix operand order repaired and checked by reading, not compiled, not run on hardware
//
#include <hip/hip_runtime.h>
#include <hip/hip_bf16.h>

// ---------------------------------------------------------------------------
// Problem constants (from reference)
// ---------------------------------------------------------------------------
#define B_  2
#define LSEQ 2048
#define DM  1024
#define DI  2048
#define NST 16
#define KCONV 4
#define RRANK 64
#define EPSV 1e-5f

#define MROWS (B_ * LSEQ)       // 4096 rows in all GEMMs

typedef __attribute__((ext_vector_type(16))) __bf16 v16bf;
typedef __attribute__((ext_vector_type(8)))  __bf16 bf16x8;
typedef __attribute__((ext_vector_type(8)))  float  v8f;

// ---------------------------------------------------------------------------
// CDNA5 async global->LDS DMA (ASYNCcnt-tracked), per CDNA5 ISA + 07.
// Each lane copies 16 contiguous bytes global -> LDS, no VGPR round trip.
// ---------------------------------------------------------------------------
__device__ __forceinline__ void async_copy_b128(uint32_t lds_off, const void* gptr) {
    asm volatile("global_load_async_to_lds_b128 %0, %1, off"
                 :: "v"(lds_off), "v"(gptr) : "memory");
}
__device__ __forceinline__ void wait_async0() {
    asm volatile("s_wait_asynccnt 0x0" ::: "memory");
}
__device__ __forceinline__ uint32_t lds_addr_of(const void* p) {
    // generic LDS pointer: flat = aperture_base(hi32) | lds_offset(lo32)
    return (uint32_t)(size_t)p;
}

// ---------------------------------------------------------------------------
// RMSNorm: per-row mean-square -> normalized bf16 output
// ---------------------------------------------------------------------------
__global__ __launch_bounds__(256) void rmsnorm_bf16_kernel(
    const float* __restrict__ x, const float* __restrict__ w,
    __bf16* __restrict__ out)
{
    const int row = blockIdx.x;
    const int tid = threadIdx.x;
    const float* xr = x + (size_t)row * DM;

    float s = 0.f;
    for (int i = tid; i < DM; i += 256) { float v = xr[i]; s += v * v; }

    __shared__ float red[256];
    red[tid] = s;
    __syncthreads();
    for (int off = 128; off > 0; off >>= 1) {
        if (tid < off) red[tid] += red[tid + off];
        __syncthreads();
    }
    const float rs = rsqrtf(red[0] / (float)DM + EPSV);

    __bf16* orow = out + (size_t)row * DM;
    for (int i = tid; i < DM; i += 256)
        orow[i] = (__bf16)(xr[i] * rs * w[i]);
}

// ---------------------------------------------------------------------------
// fp32 [K][N] -> bf16 transposed [N][K]  (weights, once per launch)
// ---------------------------------------------------------------------------
__global__ __launch_bounds__(256) void cvt_transpose_bf16_kernel(
    const float* __restrict__ src, __bf16* __restrict__ dst, int K, int N)
{
    int i = blockIdx.x * 256 + threadIdx.x;
    if (i >= N * K) return;
    const int n = i / K, k = i % K;
    dst[i] = (__bf16)src[(size_t)k * N + n];
}

// dt_r slice: first RRANK cols of xdbl [M, R+2N] -> bf16 [M, R]
__global__ __launch_bounds__(256) void slice_dtr_kernel(
    const float* __restrict__ xdbl, __bf16* __restrict__ dtr)
{
    int i = blockIdx.x * 256 + threadIdx.x;
    if (i >= MROWS * RRANK) return;
    int row = i / RRANK, c = i % RRANK;
    dtr[i] = (__bf16)xdbl[(size_t)row * (RRANK + 2 * NST) + c];
}

// ---------------------------------------------------------------------------
// Tiled bf16 WMMA GEMM: C[M,N] = A[M,K] * BT[N,K]^T (+ residual), f32 out.
// Block = 256 threads = 8 waves; tile 128x128x32; wave tile 64x32 (4x2 WMMA).
// Double-buffered LDS; tiles staged with GLOBAL_LOAD_ASYNC_TO_LDS_B128 so
// the k-step t+1 DMA overlaps the k-step t WMMAs; one barrier per k-step.
// M, K multiples of 128/32 as used here; N edge handled by FULL_N=false.
// ---------------------------------------------------------------------------
template <bool FULL_N, bool HAS_RES>
__global__ __launch_bounds__(256) void gemm_bf16_wmma_kernel(
    const __bf16* __restrict__ A,  // [M,K] row-major
    const __bf16* __restrict__ BT, // [N,K] row-major (pre-transposed weights)
    float* __restrict__ C, const float* __restrict__ residual,
    int M, int N, int K)
{
    constexpr int BM = 128, BN = 128, BK = 32;
    constexpr int LP = BK + 8;                   // pitch 40 elems = 80B (16B-aligned)
    __shared__ __bf16 As[2][BM][LP];             // [buf][m][k]
    __shared__ __bf16 Bs[2][BN][LP];             // [buf][n][k]

    const int tid  = threadIdx.x;
    const int wave = tid >> 5;
    const int lane = tid & 31;
    const int wrow = wave >> 2;                  // 0..1  -> 64-row half
    const int wcol = wave & 3;                   // 0..3  -> 32-col slice
    const int half = lane >> 4;                  // 0/1
    const int lm   = lane & 15;                  // 0..15
    const int m0   = blockIdx.y * BM;
    const int n0   = blockIdx.x * BN;

    // staging role: 2 threads per row, 16 bf16 (two b128 chunks) each
    const int sr = tid >> 1;                     // 0..127
    const int sc = (tid & 1) * 16;               // 0 or 16

    auto stage = [&](int p, int k0) {
        const __bf16* asrc = A + (size_t)(m0 + sr) * K + k0 + sc;
        async_copy_b128(lds_addr_of(&As[p][sr][sc]),     asrc);
        async_copy_b128(lds_addr_of(&As[p][sr][sc + 8]), asrc + 8);
        if (FULL_N) {
            const __bf16* bsrc = BT + (size_t)(n0 + sr) * K + k0 + sc;
            async_copy_b128(lds_addr_of(&Bs[p][sr][sc]),     bsrc);
            async_copy_b128(lds_addr_of(&Bs[p][sr][sc + 8]), bsrc + 8);
        } else {
            if (n0 + sr < N) {
                const __bf16* bsrc = BT + (size_t)(n0 + sr) * K + k0 + sc;
                *(bf16x8*)&Bs[p][sr][sc]     = *(const bf16x8*)bsrc;
                *(bf16x8*)&Bs[p][sr][sc + 8] = *(const bf16x8*)(bsrc + 8);
            } else {
                bf16x8 z = {};
                *(bf16x8*)&Bs[p][sr][sc]     = z;
                *(bf16x8*)&Bs[p][sr][sc + 8] = z;
            }
        }
    };

    v8f acc[4][2] = {};

    stage(0, 0);                       // prologue: fill buffer 0
    const int T = K / BK;
    for (int t = 0; t < T; t++) {
        const int p = t & 1;
        wait_async0();                 // buffer p's DMAs (issued at t-1) landed
        __syncthreads();               // visible to all waves; p^1 reads done
        if (t + 1 < T) stage(p ^ 1, (t + 1) * BK);   // overlap next DMA

        // ---- fragments: contiguous k = 16*half + i, same order for A and B ----
        v16bf afrag[4];
#pragma unroll
        for (int mt = 0; mt < 4; mt++) {
            const int row = wrow * 64 + mt * 16 + lm;
            bf16x8 lo = *(const bf16x8*)&As[p][row][half * 16];      // K 0..7  (+16h)
            bf16x8 hi = *(const bf16x8*)&As[p][row][half * 16 + 8];  // K 8..15 (+16h)
#pragma unroll
            for (int i = 0; i < 8; i++) { afrag[mt][i] = lo[i]; afrag[mt][8 + i] = hi[i]; }
        }
        v16bf bfrag[2];
#pragma unroll
        for (int nt = 0; nt < 2; nt++) {
            const int col = wcol * 32 + nt * 16 + lm;
            bf16x8 lo = *(const bf16x8*)&Bs[p][col][half * 16];      // K 0..7  (+16h)
            bf16x8 hi = *(const bf16x8*)&Bs[p][col][half * 16 + 8];  // K 8..15 (+16h)
#pragma unroll
            for (int i = 0; i < 8; i++) { bfrag[nt][i] = lo[i]; bfrag[nt][8 + i] = hi[i]; }
        }

#pragma unroll
        for (int mt = 0; mt < 4; mt++)
#pragma unroll
            for (int nt = 0; nt < 2; nt++)
                acc[mt][nt] = __builtin_amdgcn_wmma_f32_16x16x32_bf16(
                    false, afrag[mt], false, bfrag[nt],
                    (short)0, acc[mt][nt], false, false);
    }

    // ---- epilogue: D layout lane=N, vgpr r => M = half*8 + r ----
#pragma unroll
    for (int mt = 0; mt < 4; mt++) {
#pragma unroll
        for (int nt = 0; nt < 2; nt++) {
            const int gn = n0 + wcol * 32 + nt * 16 + lm;
            if (FULL_N || gn < N) {
#pragma unroll
                for (int r = 0; r < 8; r++) {
                    const int gm = m0 + wrow * 64 + mt * 16 + half * 8 + r;
                    float v = acc[mt][nt][r];
                    if (HAS_RES) v += residual[(size_t)gm * N + gn];
                    C[(size_t)gm * N + gn] = v;
                }
            }
        }
    }
}

// ---------------------------------------------------------------------------
// Causal depthwise conv (K=4) + bias + SiLU on the u half of xz -> bf16
// xz layout: [B*L, 2*DI], u = cols [0, DI)
// ---------------------------------------------------------------------------
__global__ __launch_bounds__(256) void conv_silu_kernel(
    const float* __restrict__ xz, const float* __restrict__ cw,
    const float* __restrict__ cb, __bf16* __restrict__ u_out)
{
    const int i = blockIdx.x * 256 + threadIdx.x;
    if (i >= MROWS * DI) return;
    const int d  = i % DI;
    const int bl = i / DI;
    const int l  = bl % LSEQ;
    const int b  = bl / LSEQ;

    float acc = cb[d];
#pragma unroll
    for (int j = 0; j < KCONV; j++) {
        const int ls = l - (KCONV - 1) + j;
        if (ls >= 0)
            acc += xz[((size_t)(b * LSEQ + ls)) * (2 * DI) + d] * cw[d * KCONV + j];
    }
    const float s = acc / (1.f + __expf(-acc));   // SiLU
    u_out[(size_t)bl * DI + d] = (__bf16)s;
}

// ---------------------------------------------------------------------------
// Selective scan: one lane per (b, d) channel; h[16] in registers.
// y = sum_n h_n*C_n + u*D ; output y * silu(z) as bf16 for the out_proj GEMM.
// ---------------------------------------------------------------------------
__global__ __launch_bounds__(256) void scan_kernel(
    const float* __restrict__ dt_in,   // [M, DI] pre-bias dt
    const float* __restrict__ dt_b,    // [DI]
    const float* __restrict__ A_log,   // [DI, N]
    const float* __restrict__ Dp,      // [DI]
    const float* __restrict__ xdbl,    // [M, R+2N] (B at +R, C at +R+N)
    const __bf16* __restrict__ u_bf,   // [M, DI]
    const float* __restrict__ xz,      // [M, 2*DI] (z at +DI)
    __bf16* __restrict__ yg)           // [M, DI]
{
    const int g = blockIdx.x * 256 + threadIdx.x;
    if (g >= B_ * DI) return;
    const int b = g / DI;
    const int d = g % DI;

    float Arow[NST];
#pragma unroll
    for (int n = 0; n < NST; n++) Arow[n] = -__expf(A_log[d * NST + n]);
    const float Dd = Dp[d];
    const float bias = dt_b[d];

    float h[NST];
#pragma unroll
    for (int n = 0; n < NST; n++) h[n] = 0.f;

    for (int l = 0; l < LSEQ; l++) {
        const size_t idx = (size_t)(b * LSEQ + l);
        const float dtp = dt_in[idx * DI + d] + bias;
        const float dtv = (dtp > 20.f) ? dtp : log1pf(__expf(dtp));
        const float u   = (float)u_bf[idx * DI + d];
        const float z   = xz[idx * (2 * DI) + DI + d];
        const float* bc = xdbl + idx * (RRANK + 2 * NST);

        float y = 0.f;
#pragma unroll
        for (int n = 0; n < NST; n++) {
            const float dA = __expf(dtv * Arow[n]);
            h[n] = dA * h[n] + (dtv * bc[RRANK + n]) * u;
            y += h[n] * bc[RRANK + NST + n];
        }
        y += u * Dd;
        const float zg = z / (1.f + __expf(-z));
        yg[idx * DI + d] = (__bf16)(y * zg);
    }
}

// ---------------------------------------------------------------------------
// Host-side orchestration
// ---------------------------------------------------------------------------
static inline size_t alignup(size_t x) { return (x + 255) & ~(size_t)255; }

extern "C" void kernel_launch(void* const* d_in, const int* in_sizes, int n_in,
                              void* d_out, int out_size, void* d_ws, size_t ws_size,
                              hipStream_t stream) {
    (void)in_sizes; (void)n_in; (void)out_size; (void)ws_size;

    const float* x         = (const float*)d_in[0];
    // d_in[1] = hormone_vectors (unused by reference)
    const float* norm_w    = (const float*)d_in[2];
    const float* in_proj_w = (const float*)d_in[3];   // [DM, 2*DI]
    const float* conv_w    = (const float*)d_in[4];   // [DI, K]
    const float* conv_b    = (const float*)d_in[5];   // [DI]
    const float* x_proj_w  = (const float*)d_in[6];   // [DI, R+2N]
    const float* dt_proj_w = (const float*)d_in[7];   // [R, DI]
    const float* dt_proj_b = (const float*)d_in[8];   // [DI]
    const float* A_log     = (const float*)d_in[9];   // [DI, N]
    const float* Dp        = (const float*)d_in[10];  // [DI]
    const float* out_proj_w= (const float*)d_in[11];  // [DI, DM]
    float* out = (float*)d_out;

    // workspace layout
    char* ws = (char*)d_ws;
    size_t off = 0;
    auto carve = [&](size_t bytes) { char* p = ws + off; off = alignup(off + bytes); return p; };

    __bf16* xn_bf    = (__bf16*)carve((size_t)MROWS * DM * 2);
    __bf16* win_bt   = (__bf16*)carve((size_t)DM * 2 * DI * 2);     // [2DI][DM]
    float*  xz       = (float*) carve((size_t)MROWS * 2 * DI * 4);
    __bf16* u_bf     = (__bf16*)carve((size_t)MROWS * DI * 2);
    __bf16* wxp_bt   = (__bf16*)carve((size_t)DI * (RRANK + 2 * NST) * 2); // [96][DI]
    float*  xdbl     = (float*) carve((size_t)MROWS * (RRANK + 2 * NST) * 4);
    __bf16* dtr_bf   = (__bf16*)carve((size_t)MROWS * RRANK * 2);
    __bf16* wdt_bt   = (__bf16*)carve((size_t)RRANK * DI * 2);      // [DI][R]
    float*  dt_in    = (float*) carve((size_t)MROWS * DI * 4);
    __bf16* yg_bf    = (__bf16*)carve((size_t)MROWS * DI * 2);
    __bf16* wout_bt  = (__bf16*)carve((size_t)DI * DM * 2);         // [DM][DI]

    // 1) rmsnorm -> bf16
    rmsnorm_bf16_kernel<<<MROWS, 256, 0, stream>>>(x, norm_w, xn_bf);

    // 2) weight conversions with transpose ([K][N] -> [N][K] bf16)
    {
        int n = DM * 2 * DI;
        cvt_transpose_bf16_kernel<<<(n + 255) / 256, 256, 0, stream>>>(
            in_proj_w, win_bt, DM, 2 * DI);
        n = DI * (RRANK + 2 * NST);
        cvt_transpose_bf16_kernel<<<(n + 255) / 256, 256, 0, stream>>>(
            x_proj_w, wxp_bt, DI, RRANK + 2 * NST);
        n = RRANK * DI;
        cvt_transpose_bf16_kernel<<<(n + 255) / 256, 256, 0, stream>>>(
            dt_proj_w, wdt_bt, RRANK, DI);
        n = DI * DM;
        cvt_transpose_bf16_kernel<<<(n + 255) / 256, 256, 0, stream>>>(
            out_proj_w, wout_bt, DI, DM);
    }

    // 3) xz = xn @ in_proj_w   [4096 x 4096 x 1024]  (N full)
    {
        dim3 grid((2 * DI) / 128, MROWS / 128);
        gemm_bf16_wmma_kernel<true, false><<<grid, 256, 0, stream>>>(
            xn_bf, win_bt, xz, nullptr, MROWS, 2 * DI, DM);
    }

    // 4) u = silu(conv(u_half)) -> bf16
    {
        int n = MROWS * DI;
        conv_silu_kernel<<<(n + 255) / 256, 256, 0, stream>>>(xz, conv_w, conv_b, u_bf);
    }

    // 5) xdbl = u @ x_proj_w   [4096 x 96 x 2048]  (N partial: one 96-wide col)
    {
        dim3 grid(1, MROWS / 128);
        gemm_bf16_wmma_kernel<false, false><<<grid, 256, 0, stream>>>(
            u_bf, wxp_bt, xdbl, nullptr, MROWS, RRANK + 2 * NST, DI);
    }

    // 6) dt_r slice -> bf16
    {
        int n = MROWS * RRANK;
        slice_dtr_kernel<<<(n + 255) / 256, 256, 0, stream>>>(xdbl, dtr_bf);
    }

    // 7) dt_in = dt_r @ dt_proj_w   [4096 x 2048 x 64]  (N full)
    {
        dim3 grid(DI / 128, MROWS / 128);
        gemm_bf16_wmma_kernel<true, false><<<grid, 256, 0, stream>>>(
            dtr_bf, wdt_bt, dt_in, nullptr, MROWS, DI, RRANK);
    }

    // 8) selective scan -> gated y (bf16)
    scan_kernel<<<(B_ * DI + 255) / 256, 256, 0, stream>>>(
        dt_in, dt_proj_b, A_log, Dp, xdbl, u_bf, xz, yg_bf);

    // 9) out = x + yg @ out_proj_w   [4096 x 1024 x 2048]  (N full + residual)
    {
        dim3 grid(DM / 128, MROWS / 128);
        gemm_bf16_wmma_kernel<true, true><<<grid, 256, 0, stream>>>(
            yg_bf, wout_bt, out, x, MROWS, DM, DI);
    }
}
